// AttentionC_21612275433490
// MI455X (gfx1250) — compile-verified, matrix operand order checked by reading, not run on hardware
//
#include <hip/hip_runtime.h>
#include <hip/hip_bf16.h>
#include <math.h>

typedef _Float16 h16;
typedef __attribute__((ext_vector_type(8)))  _Float16 v8h;
typedef __attribute__((ext_vector_type(16))) _Float16 v16h;
typedef __attribute__((ext_vector_type(8)))  float    v8f;

#define NJ    17
#define NF    32
#define CDIM  544
#define NH    8
#define HD    68
#define HDP   96    // padded head dim for Q/K (3 K-steps of 32)
#define HDP2  80    // padded head dim for O tiles (5 tiles of 16)
#define NB    8
#define NSEQ  1024
#define BN    8192  // NB*NSEQ
#define C3    1632  // 3*CDIM
#define QTPB  4     // query tiles (waves) per attention block

#define WMMA16(a,b,c) __builtin_amdgcn_wmma_f32_16x16x32_f16(false,(a),false,(b),(short)0,(c),false,false)

__device__ __forceinline__ v16h make_frag(const h16* p0, const h16* p1) {
    v8h a = *(const v8h*)p0;
    v8h b = *(const v8h*)p1;
    return __builtin_shufflevector(a, b, 0,1,2,3,4,5,6,7,8,9,10,11,12,13,14,15);
}

// ---------------- stage W: convert weights f32 -> f16 ----------------
__global__ void k_cvt(const float* __restrict__ qkvw, const float* __restrict__ projw,
                      h16* __restrict__ wq, h16* __restrict__ wp) {
    int i = blockIdx.x * blockDim.x + threadIdx.x;
    const int n1 = C3 * CDIM, n2 = CDIM * CDIM;
    if (i < n1)            wq[i] = (h16)qkvw[i];
    else if (i < n1 + n2)  wp[i - n1] = (h16)projw[i - n1];
}

// ---------------- stage Z: zero Q/K/Vt padded buffers ----------------
__global__ void k_zero(unsigned int* __restrict__ p, int nwords) {
    int i = blockIdx.x * blockDim.x + threadIdx.x;
    if (i < nwords) p[i] = 0u;
}

// ---------------- stage A: per-joint LN + pooling scores ----------------
__global__ void k_ln1(const float* __restrict__ x, const float* __restrict__ g,
                      const float* __restrict__ bb, const float* __restrict__ apw,
                      const float* __restrict__ apb,
                      float* __restrict__ xr, float* __restrict__ sc) {
    int idx = blockIdx.x * blockDim.x + threadIdx.x;  // (b*N+n)*17 + j
    if (idx >= BN * NJ) return;
    const float* px = x + (size_t)idx * NF;
    float s = 0.f, s2 = 0.f;
    #pragma unroll
    for (int f = 0; f < NF; f++) { float v = px[f]; s += v; s2 += v * v; }
    float mu = s * (1.f / NF);
    float var = s2 * (1.f / NF) - mu * mu;
    float rs = rsqrtf(var + 1e-5f);
    float* po = xr + (size_t)idx * NF;
    float acc = 0.f;
    #pragma unroll
    for (int f = 0; f < NF; f++) {
        float v = (px[f] - mu) * rs * g[f] + bb[f];
        po[f] = v;
        acc += v * apw[f];
    }
    sc[idx] = acc + apb[0];
}

// ---------------- stage B: softmax over sequence axis N ----------------
__global__ void k_softmax_n(float* __restrict__ sc) {
    int b = blockIdx.x / NJ, j = blockIdx.x % NJ;
    __shared__ float red[256];
    int t = threadIdx.x;
    float m = -3e38f;
    for (int n = t; n < NSEQ; n += 256) m = fmaxf(m, sc[(b * NSEQ + n) * NJ + j]);
    red[t] = m; __syncthreads();
    for (int o = 128; o > 0; o >>= 1) { if (t < o) red[t] = fmaxf(red[t], red[t + o]); __syncthreads(); }
    m = red[0]; __syncthreads();
    float s = 0.f;
    for (int n = t; n < NSEQ; n += 256) s += __expf(sc[(b * NSEQ + n) * NJ + j] - m);
    red[t] = s; __syncthreads();
    for (int o = 128; o > 0; o >>= 1) { if (t < o) red[t] += red[t + o]; __syncthreads(); }
    float inv = 1.f / red[0];
    for (int n = t; n < NSEQ; n += 256) {
        int id = (b * NSEQ + n) * NJ + j;
        sc[id] = __expf(sc[id] - m) * inv;
    }
}

// ---------------- stage C: pool by weights + LN over C, emit f16 ----------------
__global__ void k_pool_ln2(const float* __restrict__ xr, const float* __restrict__ w,
                           const float* __restrict__ g2, const float* __restrict__ b2,
                           h16* __restrict__ xp) {
    int row = blockIdx.x;  // b*N + n
    __shared__ float vals[CDIM];
    __shared__ float red[256];
    int t = threadIdx.x;
    float s = 0.f;
    for (int c = t; c < CDIM; c += 256) {
        float v = xr[(size_t)row * CDIM + c] * w[row * NJ + c / NF];
        vals[c] = v; s += v;
    }
    red[t] = s; __syncthreads();
    for (int o = 128; o > 0; o >>= 1) { if (t < o) red[t] += red[t + o]; __syncthreads(); }
    float mu = red[0] * (1.f / CDIM); __syncthreads();
    float s2 = 0.f;
    for (int c = t; c < CDIM; c += 256) { float d = vals[c] - mu; s2 += d * d; }
    red[t] = s2; __syncthreads();
    for (int o = 128; o > 0; o >>= 1) { if (t < o) red[t] += red[t + o]; __syncthreads(); }
    float rs = rsqrtf(red[0] * (1.f / CDIM) + 1e-5f);
    for (int c = t; c < CDIM; c += 256)
        xp[(size_t)row * CDIM + c] = (h16)((vals[c] - mu) * rs * g2[c] + b2[c]);
}

// ---------------- stage D: QKV GEMM (WMMA, 16x48 per wave) ----------------
__global__ void k_qkv_gemm(const h16* __restrict__ xp, const h16* __restrict__ wq,
                           h16* __restrict__ Qp, h16* __restrict__ Kp, h16* __restrict__ Vt) {
    int lane = threadIdx.x;                 // 0..31
    int mt = blockIdx.y * 8 + threadIdx.y;  // row tile 0..511
    int ntg = blockIdx.x;                   // col tile group 0..33 (3 tiles each)
    int r = lane & 15, hh = lane >> 4;
    const h16* arow = xp + (size_t)(mt * 16 + r) * CDIM;
    const h16* brow[3];
    #pragma unroll
    for (int nt = 0; nt < 3; nt++)
        brow[nt] = wq + (size_t)((ntg * 3 + nt) * 16 + r) * CDIM;
    v8f zero = {};
    v8f acc[3];
    #pragma unroll
    for (int nt = 0; nt < 3; nt++) acc[nt] = zero;
    for (int ks = 0; ks < CDIM; ks += 32) {
        int c0 = ks + 8 * hh, c1 = ks + 16 + 8 * hh;
        v16h a = make_frag(arow + c0, arow + c1);
        #pragma unroll
        for (int nt = 0; nt < 3; nt++) {
            v16h bfr = make_frag(brow[nt] + c0, brow[nt] + c1);
            acc[nt] = WMMA16(a, bfr, acc[nt]);
        }
    }
    const float qscale = 0.121267813f;  // 68^-0.5, folded into Q
    #pragma unroll
    for (int nt = 0; nt < 3; nt++) {
        int o = (ntg * 3 + nt) * 16 + r;
        int third = o / CDIM, rem = o % CDIM;
        int hd = rem / HD, d = rem % HD;
        #pragma unroll
        for (int vg = 0; vg < 8; vg++) {
            int m = mt * 16 + vg + 8 * hh;          // global row (b*N+n)
            int b = m >> 10, n = m & 1023;
            int bh = b * NH + hd;
            float v = acc[nt][vg];
            if (third == 0)      Qp[(size_t)(bh * NSEQ + n) * HDP + d] = (h16)(v * qscale);
            else if (third == 1) Kp[(size_t)(bh * NSEQ + n) * HDP + d] = (h16)v;
            else                 Vt[(size_t)(bh * HDP2 + d) * NSEQ + n] = (h16)v;
        }
    }
}

// ---------------- stage E: flash attention, 4 waves share K/V via LDS ----------------
__global__ void k_attn(const h16* __restrict__ Qp, const h16* __restrict__ Kp,
                       const h16* __restrict__ Vt, const float* __restrict__ sfp,
                       h16* __restrict__ O) {
    int lane = threadIdx.x;               // 0..31
    int wv = threadIdx.y;                 // 0..3 : one query tile per wave
    int tid = wv * 32 + lane;
    int qt = blockIdx.x * QTPB + wv;      // 0..63
    int hd = blockIdx.y, b = blockIdx.z;
    int bh = b * NH + hd;
    int r = lane & 15, hh = lane >> 4;
    float sf = sfp[0];

    __shared__ __align__(16) h16 kl[32 * HDP];        // 6 KB: 32 keys x 96 dims
    __shared__ __align__(16) h16 vl[HDP2 * 32];       // 5 KB: 80 dims x 32 keys
    __shared__ __align__(16) h16 p_lds[QTPB][16 * 32];// 4 KB: per-wave P tile

    // Q fragments for this wave's 16-row tile (3 K-steps covering padded 96)
    const h16* qrow = Qp + (size_t)(bh * NSEQ + qt * 16 + r) * HDP;
    v16h qa0 = make_frag(qrow +      8 * hh, qrow + 16 + 8 * hh);
    v16h qa1 = make_frag(qrow + 32 + 8 * hh, qrow + 48 + 8 * hh);
    v16h qa2 = make_frag(qrow + 64 + 8 * hh, qrow + 80 + 8 * hh);

    v8f zero = {};
    v8f accO[5];
    #pragma unroll
    for (int i = 0; i < 5; i++) accO[i] = zero;
    float m_run[8], l_run[8];
    #pragma unroll
    for (int i = 0; i < 8; i++) { m_run[i] = -1e30f; l_run[i] = 0.f; }

    const h16* kbase = Kp + (size_t)bh * NSEQ * HDP;
    const h16* vbase = Vt + (size_t)bh * HDP2 * NSEQ;
    const float invN1 = 1.f / (float)(NSEQ - 1);

    for (int kt = 0; kt < NSEQ; kt += 32) {
        // ---- cooperative staging: K tile is contiguous (keys are rows) ----
        __syncthreads();  // previous iteration's LDS reads complete
        {
            const v8h* kg = (const v8h*)(kbase + (size_t)kt * HDP);
            v8h* kls = (v8h*)kl;
            #pragma unroll
            for (int i = 0; i < 3; i++)           // 32*96/8 = 384 chunks
                kls[tid + 128 * i] = kg[tid + 128 * i];
            v8h* vls = (v8h*)vl;
            for (int i = tid; i < HDP2 * 4; i += 128) {  // 80 rows x 4 chunks
                int row = i >> 2, c = i & 3;
                vls[i] = *(const v8h*)(vbase + (size_t)row * NSEQ + kt + c * 8);
            }
        }
        __syncthreads();

        // ---- S = Q K^T for two 16-key halves (fragments from LDS) ----
        v8f s0 = zero, s1 = zero;
        const h16* krow0 = kl + (size_t)r * HDP;
        const h16* krow1 = kl + (size_t)(16 + r) * HDP;
        v16h kb;
        kb = make_frag(krow0 +      8 * hh, krow0 + 16 + 8 * hh); s0 = WMMA16(qa0, kb, s0);
        kb = make_frag(krow1 +      8 * hh, krow1 + 16 + 8 * hh); s1 = WMMA16(qa0, kb, s1);
        kb = make_frag(krow0 + 32 + 8 * hh, krow0 + 48 + 8 * hh); s0 = WMMA16(qa1, kb, s0);
        kb = make_frag(krow1 + 32 + 8 * hh, krow1 + 48 + 8 * hh); s1 = WMMA16(qa1, kb, s1);
        kb = make_frag(krow0 + 64 + 8 * hh, krow0 + 80 + 8 * hh); s0 = WMMA16(qa2, kb, s0);
        kb = make_frag(krow1 + 64 + 8 * hh, krow1 + 80 + 8 * hh); s1 = WMMA16(qa2, kb, s1);

        // per-key positional scale (lane == key column in C-layout)
        float p0 = (float)(kt + r) * invN1 - 0.5f;
        float p1 = (float)(kt + 16 + r) * invN1 - 0.5f;
        float ps0 = __expf(-sf * p0 * p0);
        float ps1 = __expf(-sf * p1 * p1);

        // ---- online softmax, per query row (row = vg + 8*hh) ----
        float pv0[8], pv1[8];
        #pragma unroll
        for (int vg = 0; vg < 8; vg++) {
            float t0 = s0[vg] * ps0;
            float t1 = s1[vg] * ps1;
            float tm = fmaxf(t0, t1);
            #pragma unroll
            for (int msk = 8; msk >= 1; msk >>= 1)
                tm = fmaxf(tm, __shfl_xor(tm, msk, 32));
            float mn = fmaxf(m_run[vg], tm);
            float alpha = __expf(m_run[vg] - mn);
            float e0 = __expf(t0 - mn), e1 = __expf(t1 - mn);
            float es = e0 + e1;
            #pragma unroll
            for (int msk = 8; msk >= 1; msk >>= 1)
                es += __shfl_xor(es, msk, 32);
            l_run[vg] = l_run[vg] * alpha + es;
            m_run[vg] = mn;
            #pragma unroll
            for (int dt = 0; dt < 5; dt++) accO[dt][vg] *= alpha;
            pv0[vg] = e0; pv1[vg] = e1;
        }

        // ---- stash P tile (16x32) in this wave's LDS slice, reload as A-frag ----
        h16* pw = p_lds[wv];
        #pragma unroll
        for (int vg = 0; vg < 8; vg++) {
            int row = vg + 8 * hh;
            pw[row * 32 + r]      = (h16)pv0[vg];
            pw[row * 32 + 16 + r] = (h16)pv1[vg];
        }
        const h16* prow = pw + r * 32;       // same-wave DS ops are in-order
        v16h pf = make_frag(prow + 8 * hh, prow + 16 + 8 * hh);

        // ---- O += P @ V  (V fragments from LDS, key-contiguous rows of 32) ----
        #pragma unroll
        for (int dt = 0; dt < 5; dt++) {
            const h16* vrow = vl + (dt * 16 + r) * 32;
            v16h vf = make_frag(vrow + 8 * hh, vrow + 16 + 8 * hh);
            accO[dt] = WMMA16(pf, vf, accO[dt]);
        }
    }

    // ---- finalize: normalize and scatter to O[b,n, h*68+d] ----
    #pragma unroll
    for (int vg = 0; vg < 8; vg++) {
        float inv = 1.f / l_run[vg];
        int q = qt * 16 + vg + 8 * hh;
        #pragma unroll
        for (int dt = 0; dt < 5; dt++) {
            int d = dt * 16 + r;
            if (d < HD)
                O[(size_t)(b * NSEQ + q) * CDIM + hd * HD + d] = (h16)(accO[dt][vg] * inv);
        }
    }
}

// ---------------- stage F: projection GEMM (WMMA) + bias ----------------
__global__ void k_proj(const h16* __restrict__ O, const h16* __restrict__ wp,
                       const float* __restrict__ pb, float* __restrict__ out) {
    int lane = threadIdx.x;
    int mt = blockIdx.y * 8 + threadIdx.y;  // 0..511
    int ntg = blockIdx.x;                   // 0..16 (2 tiles each -> 34 tiles = 544)
    int r = lane & 15, hh = lane >> 4;
    const h16* arow = O + (size_t)(mt * 16 + r) * CDIM;
    int o0 = (ntg * 2 + 0) * 16 + r;
    int o1 = (ntg * 2 + 1) * 16 + r;
    const h16* b0row = wp + (size_t)o0 * CDIM;
    const h16* b1row = wp + (size_t)o1 * CDIM;
    v8f acc0 = {}; v8f acc1 = {};
    for (int ks = 0; ks < CDIM; ks += 32) {
        int c0 = ks + 8 * hh, c1 = ks + 16 + 8 * hh;
        v16h a  = make_frag(arow  + c0, arow  + c1);
        v16h b0 = make_frag(b0row + c0, b0row + c1);
        v16h b1 = make_frag(b1row + c0, b1row + c1);
        acc0 = WMMA16(a, b0, acc0);
        acc1 = WMMA16(a, b1, acc1);
    }
    float bias0 = pb[o0], bias1 = pb[o1];
    #pragma unroll
    for (int vg = 0; vg < 8; vg++) {
        int m = mt * 16 + vg + 8 * hh;
        out[(size_t)m * CDIM + o0] = acc0[vg] + bias0;
        out[(size_t)m * CDIM + o1] = acc1[vg] + bias1;
    }
}

extern "C" void kernel_launch(void* const* d_in, const int* in_sizes, int n_in,
                              void* d_out, int out_size, void* d_ws, size_t ws_size,
                              hipStream_t stream) {
    const float* x     = (const float*)d_in[0];
    const float* ng    = (const float*)d_in[1];
    const float* nb    = (const float*)d_in[2];
    const float* apw   = (const float*)d_in[3];
    const float* apb   = (const float*)d_in[4];
    const float* n2g   = (const float*)d_in[5];
    const float* n2b   = (const float*)d_in[6];
    const float* qkvw  = (const float*)d_in[7];
    const float* projw = (const float*)d_in[8];
    const float* projb = (const float*)d_in[9];
    const float* sfp   = (const float*)d_in[10];
    float* out = (float*)d_out;

    char* ws = (char*)d_ws;
    size_t off = 0;
    float* xr = (float*)(ws + off); off += (size_t)BN * CDIM * 4;        // 17.8 MB
    float* sc = (float*)(ws + off); off += (size_t)BN * NJ * 4;          // 0.56 MB
    h16* xp = (h16*)(ws + off); off += (size_t)BN * CDIM * 2;            // 8.9 MB
    h16* wq = (h16*)(ws + off); off += (size_t)C3 * CDIM * 2;            // 1.8 MB
    h16* wp = (h16*)(ws + off); off += (size_t)CDIM * CDIM * 2;          // 0.6 MB
    h16* Qp = (h16*)(ws + off); off += (size_t)NB * NH * NSEQ * HDP * 2; // 12.6 MB
    h16* Kp = (h16*)(ws + off); off += (size_t)NB * NH * NSEQ * HDP * 2; // 12.6 MB
    h16* Vt = (h16*)(ws + off); off += (size_t)NB * NH * HDP2 * NSEQ * 2;// 10.5 MB
    h16* O  = (h16*)(ws + off); off += (size_t)BN * CDIM * 2;            // 8.9 MB

    // W: weight conversion f32 -> f16
    int ncvt = C3 * CDIM + CDIM * CDIM;
    k_cvt<<<(ncvt + 255) / 256, 256, 0, stream>>>(qkvw, projw, wq, wp);
    // Z: zero padded Q/K/Vt region (contiguous)
    int zwords = (int)(((size_t)NB * NH * NSEQ * HDP * 2 * 2 +
                        (size_t)NB * NH * HDP2 * NSEQ * 2) / 4);
    k_zero<<<(zwords + 255) / 256, 256, 0, stream>>>((unsigned int*)Qp, zwords);
    // A: per-joint LN + attention-pool scores
    k_ln1<<<(BN * NJ + 255) / 256, 256, 0, stream>>>(x, ng, nb, apw, apb, xr, sc);
    // B: softmax over sequence axis
    k_softmax_n<<<NB * NJ, 256, 0, stream>>>(sc);
    // C: pool + LN over C -> f16 activations
    k_pool_ln2<<<BN, 256, 0, stream>>>(xr, sc, n2g, n2b, xp);
    // D: QKV GEMM (WMMA), scatter to Q (scaled), K, V^T
    k_qkv_gemm<<<dim3(34, 64), dim3(32, 8), 0, stream>>>(xp, wq, Qp, Kp, Vt);
    // E: flash attention, 4 query tiles per block sharing K/V through LDS
    k_attn<<<dim3(64 / QTPB, NH, NB), dim3(32, QTPB), 0, stream>>>(Qp, Kp, Vt, sfp, O);
    // F: projection GEMM (WMMA) + bias -> f32 output
    k_proj<<<dim3(17, 64), dim3(32, 8), 0, stream>>>(O, wp, projb, out);
    (void)in_sizes; (void)n_in; (void)out_size; (void)ws_size;
}
